// Encoder_30897994727668
// MI455X (gfx1250) — matrix operand order repaired and checked by reading, not compiled, not run on hardware
//
#include <hip/hip_runtime.h>
#include <hip/hip_bf16.h>

// ---------------------------------------------------------------------------
// Fused "stateless-LSTM" kernel for MI455X (gfx1250, wave32, WMMA).
//
// gates = x @ W_ih^T + (b_ih + b_hh); f-gate is dead (c_prev == 0), so only
// the i / g / o gate columns are computed:
//   c = sigmoid(i) * tanh(g);  h = sigmoid(o) * tanh(c)
//
// M = 131072 rows, K = 128, N_eff = 768 (3 gates x 256).
// Memory-bound: ~335 MB moved -> ~14.4 us floor at 23.3 TB/s. GEMM runs on
// v_wmma_f32_16x16x32_f16 (f32 accumulate); activations use v_rcp/v_exp
// (or hardware v_tanh if the builtin exists) to keep VALU issue below the
// bandwidth floor -- plain f32 division would lower to the ~10-op IEEE
// div sequence and make instruction issue the bottleneck.
// ---------------------------------------------------------------------------

typedef __attribute__((ext_vector_type(16))) _Float16 v16h;
typedef __attribute__((ext_vector_type(8)))  _Float16 v8h;
typedef __attribute__((ext_vector_type(8)))  float    v8f;
typedef __attribute__((ext_vector_type(4)))  float    v4f;

#define K_DIM    128
#define HID      256
#define NROWS    (64 * 2048)                 /* 131072 */
#define H_ELEMS  ((long)NROWS * HID)         /* 33554432 */
#define LDS_PITCH 136                        /* 128 + 8 halves: 272B row stride, conflict-free */

// ws layout: [0, 256KB) f16 W_ih (1024x128), then 1024 f32 combined bias.
#define WS_WH_ELEMS (1024 * 128)

// ---- prep: W_ih -> f16, bias = b_ih + b_hh --------------------------------
__global__ void lstm_prep_kernel(const float* __restrict__ W,
                                 const float* __restrict__ b_ih,
                                 const float* __restrict__ b_hh,
                                 _Float16* __restrict__ wh,
                                 float* __restrict__ bias) {
    int idx = blockIdx.x * 256 + threadIdx.x;
    if (idx < WS_WH_ELEMS) wh[idx] = (_Float16)W[idx];
    if (idx < 1024)        bias[idx] = b_ih[idx] + b_hh[idx];
}

// ---- fast activations: single-instruction trans ops only ------------------
#if __has_builtin(__builtin_amdgcn_tanhf)
#define HAVE_HW_TANH 1
__device__ __forceinline__ float fast_tanh(float v) {
    return __builtin_amdgcn_tanhf(v);          // v_tanh_f32
}
#elif __has_builtin(__builtin_amdgcn_tanh_f32)
#define HAVE_HW_TANH 1
__device__ __forceinline__ float fast_tanh(float v) {
    return __builtin_amdgcn_tanh_f32(v);       // v_tanh_f32
}
#else
#define HAVE_HW_TANH 0
__device__ __forceinline__ float fast_tanh(float v) {
    // tanh(x) = 1 - 2/(exp(2x)+1): mul, v_exp, add, v_rcp, fma
    return __builtin_fmaf(-2.0f,
        __builtin_amdgcn_rcpf(__expf(2.0f * v) + 1.0f), 1.0f);
}
#endif

__device__ __forceinline__ float fast_sigmoid(float v) {
#if HAVE_HW_TANH
    // sigmoid(x) = 0.5 + 0.5*tanh(x/2): mul, v_tanh, fma
    return __builtin_fmaf(0.5f, fast_tanh(0.5f * v), 0.5f);
#else
    // mul, v_exp, add, v_rcp
    return __builtin_amdgcn_rcpf(1.0f + __expf(-v));
#endif
}

// ---- main fused GEMM + activation kernel ----------------------------------
// Block = 256 threads (8 wave32). Block owns one 16-row M-tile (staged in LDS
// as f16). Each wave computes a 16x16 tile of hidden index j, accumulating the
// i/g/o gate GEMMs with v_wmma_f32_16x16x32_f16, then fuses the activations.
__global__ __launch_bounds__(256) void lstm_wmma_kernel(
        const float* __restrict__ x,
        const _Float16* __restrict__ wh,
        const float* __restrict__ bias,
        float* __restrict__ out) {
    __shared__ __align__(16) _Float16 xs[16 * LDS_PITCH];

    const int tid  = threadIdx.x;
    const int lane = tid & 31;
    const int wave = tid >> 5;

    const int  mtile = blockIdx.x >> 1;                // 0..8191
    const int  jtile = ((blockIdx.x & 1) << 3) + wave; // 0..15
    const long m0    = (long)mtile * 16;

    // ---- stage x tile (16 rows x 128 f32) into LDS as f16 -----------------
    {
        const int r = tid >> 4;          // row 0..15
        const int c = (tid & 15) * 8;    // col 0,8,...,120
        const float* src = x + (m0 + r) * K_DIM + c;
        v4f f0 = *(const v4f*)(src);
        v4f f1 = *(const v4f*)(src + 4);
        v8h p;
#pragma unroll
        for (int t = 0; t < 4; ++t) { p[t] = (_Float16)f0[t]; p[4 + t] = (_Float16)f1[t]; }
        *(v8h*)(&xs[r * LDS_PITCH + c]) = p;
    }
    __syncthreads();

    // ---- per-lane WMMA operand addressing (16-bit A 16x32 / B Kx16, contiguous k = 16*grp + s) ---
    const int col = lane & 15;   // N column within tile / A row
    const int grp = lane >> 4;   // half-wave group
    const int jc  = (jtile << 4) + col;          // hidden index 0..255

    const _Float16* __restrict__ Bi = wh + (long)(jc)       * K_DIM; // i-gate row
    const _Float16* __restrict__ Bg = wh + (long)(512 + jc) * K_DIM; // g-gate row
    const _Float16* __restrict__ Bo = wh + (long)(768 + jc) * K_DIM; // o-gate row
    const _Float16* __restrict__ Arow = &xs[col * LDS_PITCH];

    v8f acc_i = {}; v8f acc_g = {}; v8f acc_o = {};

#pragma unroll
    for (int ks = 0; ks < 4; ++ks) {
        const int kb = ks * 32;
        // A layout: lanes g=0: halves 0..15 = K kb+0..15
        //           lanes g=1: halves 0..15 = K kb+16..31
        v8h lo = *(const v8h*)(Arow + kb + grp * 16);
        v8h hi = *(const v8h*)(Arow + kb + grp * 16 + 8);
        v16h a = __builtin_shufflevector(lo, hi,
                 0, 1, 2, 3, 4, 5, 6, 7, 8, 9, 10, 11, 12, 13, 14, 15);

        // B layout: lane holds 16 consecutive K for its column:
        //           g=0 -> K kb+0..15, g=1 -> K kb+16..31   (32B, aligned)
        v16h bi = *(const v16h*)(Bi + kb + grp * 16);
        v16h bg = *(const v16h*)(Bg + kb + grp * 16);
        v16h bo = *(const v16h*)(Bo + kb + grp * 16);

        acc_i = __builtin_amdgcn_wmma_f32_16x16x32_f16(false, a, false, bi,
                                                       (short)0, acc_i, false, false);
        acc_g = __builtin_amdgcn_wmma_f32_16x16x32_f16(false, a, false, bg,
                                                       (short)0, acc_g, false, false);
        acc_o = __builtin_amdgcn_wmma_f32_16x16x32_f16(false, a, false, bo,
                                                       (short)0, acc_o, false, false);
    }

    // ---- fused activation epilogue + nontemporal streaming stores ---------
    const float bI = bias[jc];
    const float bG = bias[512 + jc];
    const float bO = bias[768 + jc];

    // Base pointers once; the per-v offsets (v*HID floats = 1KB) fold into
    // the 24-bit instruction immediate.
    float* __restrict__ outh = out +           (m0 + grp * 8) * HID + jc;
    float* __restrict__ outc = out + H_ELEMS + (m0 + grp * 8) * HID + jc;

#pragma unroll
    for (int v = 0; v < 8; ++v) {
        // C/D layout: acc[v] of lane L -> row m0 + v + (L/16)*8, col jc
        const float gi = acc_i[v] + bI;
        const float gg = acc_g[v] + bG;
        const float go = acc_o[v] + bO;
        const float c  = fast_sigmoid(gi) * fast_tanh(gg);
        const float h  = fast_sigmoid(go) * fast_tanh(c);
        __builtin_nontemporal_store(h, outh + (long)v * HID);
        __builtin_nontemporal_store(c, outc + (long)v * HID);
    }
}

extern "C" void kernel_launch(void* const* d_in, const int* in_sizes, int n_in,
                              void* d_out, int out_size, void* d_ws, size_t ws_size,
                              hipStream_t stream) {
    (void)in_sizes; (void)n_in; (void)out_size; (void)ws_size;
    const float* x    = (const float*)d_in[0];
    const float* W_ih = (const float*)d_in[1];
    // d_in[2] = W_hh: provably unused (h_prev == 0 for every step)
    const float* b_ih = (const float*)d_in[3];
    const float* b_hh = (const float*)d_in[4];

    _Float16* wh   = (_Float16*)d_ws;
    float*    bias = (float*)((char*)d_ws + (size_t)WS_WH_ELEMS * sizeof(_Float16));

    // 512 * 256 = 131072 threads covers W_ih conversion + bias fold.
    lstm_prep_kernel<<<512, 256, 0, stream>>>(W_ih, b_ih, b_hh, wh, bias);

    // 8192 M-tiles * 2 blocks per tile (8 j-tiles each) = 16384 blocks.
    lstm_wmma_kernel<<<16384, 256, 0, stream>>>(x, wh, bias, (float*)d_out);
}
